// Model_1537598292238
// MI455X (gfx1250) — matrix operand order repaired and checked by reading, not compiled, not run on hardware
//
#include <hip/hip_runtime.h>

// CDNA5 / gfx1250, wave32. Matrix math: v_wmma_f32_16x16x32_f16.
// Hot-loop LDS staging uses async global->LDS (ASYNCcnt) where it is a
// straight copy; transposes/format conversions are hoisted into tiny
// one-shot prep kernels so the GEMM/attention inner loops stay clean.

typedef __attribute__((ext_vector_type(16))) _Float16 v16h;
typedef __attribute__((ext_vector_type(8)))  _Float16 h8;
typedef __attribute__((ext_vector_type(8)))  float    v8f;
typedef __attribute__((ext_vector_type(4)))  float    f4;

union V16 { v16h v; h8 h[2]; };

static __device__ __forceinline__ v8f wmma_f16(v16h a, v16h b, v8f c) {
  return __builtin_amdgcn_wmma_f32_16x16x32_f16(false, a, false, b, (short)0, c,
                                                false, false);
}

// Async 16B global -> LDS copy (per lane). VDST = LDS byte address VGPR,
// GV addressing (64-bit VGPR address, no SADDR). Tracked by ASYNCcnt.
static __device__ __forceinline__ void async_g2l_b128(void* lds, const void* g) {
  unsigned l = (unsigned)(size_t)lds;  // generic->LDS: low 32 bits are the LDS addr
  asm volatile("global_load_async_to_lds_b128 %0, %1, off"
               :: "v"(l), "v"(g) : "memory");
}
static __device__ __forceinline__ void wait_async0() {
  asm volatile("s_wait_asynccnt 0x0" ::: "memory");
}

// ---------------------------------------------------------------------------
// One-shot prep: Wt[n][k] = (f16)W[k][n], 1024x1024, LDS-tiled 64x64.
// ---------------------------------------------------------------------------
__global__ __launch_bounds__(256) void wtrans(const float* __restrict__ W,
                                              _Float16* __restrict__ Wt) {
  __shared__ _Float16 T[64 * 72];  // T[n_local][k_local]
  const int k0 = blockIdx.x * 64, n0 = blockIdx.y * 64;
  const int tid = threadIdx.x;
  const int r = tid >> 2, s = (tid & 3) * 16;
  const f4* src = (const f4*)(W + (size_t)(k0 + r) * 1024 + n0 + s);
  f4 a0 = src[0], a1 = src[1], a2 = src[2], a3 = src[3];
#pragma unroll
  for (int j = 0; j < 4; ++j) {
    T[(s + j)      * 72 + r] = (_Float16)a0[j];
    T[(s + 4 + j)  * 72 + r] = (_Float16)a1[j];
    T[(s + 8 + j)  * 72 + r] = (_Float16)a2[j];
    T[(s + 12 + j) * 72 + r] = (_Float16)a3[j];
  }
  __syncthreads();
  h8* dst = (h8*)(Wt + (size_t)(n0 + r) * 1024 + k0 + s);
  dst[0] = *(const h8*)&T[r * 72 + s];
  dst[1] = *(const h8*)&T[r * 72 + s + 8];
}

// ---------------------------------------------------------------------------
// One-shot prep: Vt[hb][d][key] = V[hb][key][d] (f16), per-hb 1024x64 tiles.
// ---------------------------------------------------------------------------
__global__ __launch_bounds__(256) void vtrans(const _Float16* __restrict__ V,
                                              _Float16* __restrict__ Vt) {
  __shared__ _Float16 T[64 * 72];  // T[d][key_local]
  const int key0 = blockIdx.x * 64;
  const size_t base = (size_t)blockIdx.y * 65536;
  const int tid = threadIdx.x;
  const int r = tid >> 2, s = (tid & 3) * 16;
  const h8* src = (const h8*)(V + base + (size_t)(key0 + r) * 64 + s);
  h8 x0 = src[0], x1 = src[1];
#pragma unroll
  for (int j = 0; j < 8; ++j) {
    T[(s + j) * 72 + r]     = x0[j];
    T[(s + 8 + j) * 72 + r] = x1[j];
  }
  __syncthreads();
  h8* dst = (h8*)(Vt + base + (size_t)r * 1024 + key0 + s);
  dst[0] = *(const h8*)&T[r * 72 + s];
  dst[1] = *(const h8*)&T[r * 72 + s + 8];
}

// ---------------------------------------------------------------------------
// GEMM: C[8192,1024] = A[8192,1024] @ W[1024,1024] + bias, W given as
// pre-transposed f16 Wt[n][k]. WG = 8 waves, tile 128x128, K-step 64.
// B (and A when already f16) staged via async global->LDS b128 copies.
// ---------------------------------------------------------------------------
template <typename AIN, typename COUT>
__global__ __launch_bounds__(256) void gemm128(const AIN* __restrict__ A,
                                               const _Float16* __restrict__ Wt,
                                               const float* __restrict__ bias,
                                               COUT* __restrict__ C) {
  constexpr int KD = 1024, ND = 1024, TK = 64;
  constexpr int LDA = 72;  // halves per LDS row (144B, 16B-aligned, padded)

  __shared__ _Float16 Asm[128 * LDA];
  __shared__ _Float16 Bsm[128 * LDA];  // Bsm[n][k]

  const int tid  = threadIdx.x;
  const int lane = tid & 31;
  const int wv   = tid >> 5;
  const int lh   = lane & 15;
  const bool lo  = lane < 16;

  const int blockN = blockIdx.x * 128;
  const int blockM = blockIdx.y * 128;
  const int wm = (wv & 3) * 32;
  const int wn = (wv >> 2) * 64;

  v8f acc[2][4];
#pragma unroll
  for (int mi = 0; mi < 2; ++mi)
#pragma unroll
    for (int t = 0; t < 4; ++t) acc[mi][t] = {};

  const int srow = tid >> 1;           // 0..127
  const int sseg = (tid & 1) * 32;     // element offset (halves/floats)

  for (int k0 = 0; k0 < KD; k0 += TK) {
    // stage A tile 128x64 -> f16 LDS
    if constexpr (sizeof(AIN) == 4) {
      const f4* s4 = (const f4*)(A + (size_t)(blockM + srow) * KD + k0 + sseg);
      _Float16* dst = &Asm[srow * LDA + sseg];
#pragma unroll
      for (int q = 0; q < 4; ++q) {
        f4 a = s4[2 * q], b = s4[2 * q + 1];
        h8 y;
#pragma unroll
        for (int j = 0; j < 4; ++j) { y[j] = (_Float16)a[j]; y[4 + j] = (_Float16)b[j]; }
        ((h8*)dst)[q] = y;
      }
    } else {
      const char* src = (const char*)(A + (size_t)(blockM + srow) * KD + k0 + sseg);
      char* dst = (char*)&Asm[srow * LDA + sseg];
#pragma unroll
      for (int q = 0; q < 4; ++q) async_g2l_b128(dst + 16 * q, src + 16 * q);
    }
    // stage B tile 128(n) x 64(k): straight copy from Wt
    {
      const char* src = (const char*)(Wt + (size_t)(blockN + srow) * KD + k0 + sseg);
      char* dst = (char*)&Bsm[srow * LDA + sseg];
#pragma unroll
      for (int q = 0; q < 4; ++q) async_g2l_b128(dst + 16 * q, src + 16 * q);
    }
    wait_async0();
    __syncthreads();

#pragma unroll
    for (int i = 0; i < 2; ++i) {  // K sub-steps of 32
      V16 af[2];
#pragma unroll
      for (int mi = 0; mi < 2; ++mi) {
        const _Float16* p = &Asm[(wm + mi * 16 + lh) * LDA + i * 32 + (lo ? 0 : 8)];
        af[mi].h[0] = *(const h8*)p;
        af[mi].h[1] = *(const h8*)(p + 16);
      }
#pragma unroll
      for (int t = 0; t < 4; ++t) {
        const _Float16* p = &Bsm[(wn + t * 16 + lh) * LDA + i * 32 + (lo ? 0 : 8)];
        V16 bf;
        bf.h[0] = *(const h8*)p;
        bf.h[1] = *(const h8*)(p + 16);
#pragma unroll
        for (int mi = 0; mi < 2; ++mi)
          acc[mi][t] = wmma_f16(af[mi].v, bf.v, acc[mi][t]);
      }
    }
    __syncthreads();
  }

#pragma unroll
  for (int mi = 0; mi < 2; ++mi) {
    const int row0 = blockM + wm + mi * 16 + (lo ? 0 : 8);
#pragma unroll
    for (int t = 0; t < 4; ++t) {
      const int col = blockN + wn + t * 16 + lh;
      const float b = bias[col];
#pragma unroll
      for (int r = 0; r < 8; ++r)
        C[(size_t)(row0 + r) * ND + col] = (COUT)(acc[mi][t][r] + b);
    }
  }
}

// ---------------------------------------------------------------------------
// Attention per (head*batch): Q',K' contiguous [1024x64] f16, V pre-transposed
// Vt[hb][d][key]. WG = 8 waves = 128 query rows. Two passes over 64-key
// chunks: pass0 row max+mean, pass1 masked exp / denom / P@V.
// ---------------------------------------------------------------------------
__global__ __launch_bounds__(256) void attn128(const _Float16* __restrict__ Qb,
                                               const _Float16* __restrict__ Kb,
                                               const _Float16* __restrict__ Vtb,
                                               _Float16* __restrict__ Ob) {
  constexpr int L = 1024, D = 64, LDK = 72;
  constexpr float SCALE = 0.03125f;  // 1/sqrt(1024)

  __shared__ _Float16 Ksm[64 * LDK];      // [key][dim]
  __shared__ _Float16 Vtm[64 * LDK];      // [dim][key]
  __shared__ _Float16 Psm[8 * 16 * LDK];  // per-wave probs [row][key]

  const int hb = blockIdx.x;
  const size_t base = (size_t)hb * L * D;
  const int tid = threadIdx.x, lane = tid & 31, wv = tid >> 5;
  const int lh = lane & 15;
  const bool lo = lane < 16;
  const int qrow = blockIdx.y * 128 + wv * 16;

  V16 qf[2];
  {
    const _Float16* qp = Qb + base + (size_t)(qrow + lh) * D;
#pragma unroll
    for (int i = 0; i < 2; ++i) {
      const _Float16* p = qp + i * 32 + (lo ? 0 : 8);
      qf[i].h[0] = *(const h8*)p;
      qf[i].h[1] = *(const h8*)(p + 16);
    }
  }

  float rmax[8], rsum[8], mean[8], dsum[8];
#pragma unroll
  for (int r = 0; r < 8; ++r) {
    rmax[r] = -3.0e38f; rsum[r] = 0.f; dsum[r] = 0.f; mean[r] = 0.f;
  }
  v8f o[4];
#pragma unroll
  for (int t = 0; t < 4; ++t) o[t] = {};

  const int ck = tid >> 2;        // staging row 0..63
  const int cs = (tid & 3) * 16;  // staging segment
  _Float16* psw = &Psm[wv * 16 * LDK];

  for (int pass = 0; pass < 2; ++pass) {
    for (int key0 = 0; key0 < L; key0 += 64) {
      {  // K chunk: straight async copy [key][dim]
        const char* src = (const char*)(Kb + base + (size_t)(key0 + ck) * D + cs);
        char* dst = (char*)&Ksm[ck * LDK + cs];
        async_g2l_b128(dst, src);
        async_g2l_b128(dst + 16, src + 16);
      }
      if (pass == 1) {  // Vt chunk: straight async copy [dim][key]
        const char* src = (const char*)(Vtb + base + (size_t)ck * 1024 + key0 + cs);
        char* dst = (char*)&Vtm[ck * LDK + cs];
        async_g2l_b128(dst, src);
        async_g2l_b128(dst + 16, src + 16);
      }
      wait_async0();
      __syncthreads();

#pragma unroll
      for (int t = 0; t < 4; ++t) {
        v8f s = {};
#pragma unroll
        for (int i = 0; i < 2; ++i) {
          const _Float16* p = &Ksm[(t * 16 + lh) * LDK + i * 32 + (lo ? 0 : 8)];
          V16 bf;
          bf.h[0] = *(const h8*)p;
          bf.h[1] = *(const h8*)(p + 16);
          s = wmma_f16(qf[i].v, bf.v, s);
        }
        if (pass == 0) {
#pragma unroll
          for (int r = 0; r < 8; ++r) {
            float v = s[r] * SCALE;
            rmax[r] = fmaxf(rmax[r], v);
            rsum[r] += v;
          }
        } else {
#pragma unroll
          for (int r = 0; r < 8; ++r) {
            float v = s[r] * SCALE;
            float pr = (v > mean[r]) ? __expf(v - rmax[r]) : 0.0f;  // strict >
            dsum[r] += pr;
            psw[(r + (lo ? 0 : 8)) * LDK + t * 16 + lh] = (_Float16)pr;
          }
        }
      }

      if (pass == 1) {
        asm volatile("s_wait_dscnt 0" ::: "memory");  // LDS in-order per wave
#pragma unroll
        for (int i = 0; i < 2; ++i) {
          const _Float16* ap = &psw[lh * LDK + i * 32 + (lo ? 0 : 8)];
          V16 af;
          af.h[0] = *(const h8*)ap;
          af.h[1] = *(const h8*)(ap + 16);
#pragma unroll
          for (int t = 0; t < 4; ++t) {
            const _Float16* vp = &Vtm[(t * 16 + lh) * LDK + i * 32 + (lo ? 0 : 8)];
            V16 bf;
            bf.h[0] = *(const h8*)vp;
            bf.h[1] = *(const h8*)(vp + 16);
            o[t] = wmma_f16(af.v, bf.v, o[t]);
          }
        }
      }
      __syncthreads();
    }
    if (pass == 0) {  // 16-lane half reductions match C-layout row halves
#pragma unroll
      for (int r = 0; r < 8; ++r) {
#pragma unroll
        for (int m = 1; m < 16; m <<= 1) {
          rsum[r] += __shfl_xor(rsum[r], m, 32);
          rmax[r] = fmaxf(rmax[r], __shfl_xor(rmax[r], m, 32));
        }
        mean[r] = rsum[r] * (1.0f / 1024.0f);
      }
    }
  }

#pragma unroll
  for (int r = 0; r < 8; ++r) {
#pragma unroll
    for (int m = 1; m < 16; m <<= 1) dsum[r] += __shfl_xor(dsum[r], m, 32);
    dsum[r] = (dsum[r] > 0.f) ? (1.0f / dsum[r]) : 0.f;
  }

#pragma unroll
  for (int t = 0; t < 4; ++t) {
#pragma unroll
    for (int r = 0; r < 8; ++r) {
      const int row = qrow + (lo ? 0 : 8) + r;
      Ob[base + (size_t)row * D + t * 16 + lh] = (_Float16)(o[t][r] * dsum[r]);
    }
  }
}

// ---------------------------------------------------------------------------
// Launch. Workspace (f16): Q,K,V,O (16 MiB each) + Vt (16 MiB) + Wt (2 MiB).
// Wt buffer is reused per projection (same stream serializes).
// ---------------------------------------------------------------------------
extern "C" void kernel_launch(void* const* d_in, const int* in_sizes, int n_in,
                              void* d_out, int out_size, void* d_ws,
                              size_t ws_size, hipStream_t stream) {
  (void)in_sizes; (void)n_in; (void)out_size; (void)ws_size;
  const float* x  = (const float*)d_in[0];
  const float* y  = (const float*)d_in[1];
  const float* Wq = (const float*)d_in[2];
  const float* bq = (const float*)d_in[3];
  const float* Wk = (const float*)d_in[4];
  const float* bk = (const float*)d_in[5];
  const float* Wv = (const float*)d_in[6];
  const float* bv = (const float*)d_in[7];
  const float* Wo = (const float*)d_in[8];
  const float* bo = (const float*)d_in[9];

  const size_t SZ = (size_t)8192 * 1024;
  _Float16* Qh  = (_Float16*)d_ws;
  _Float16* Kh  = Qh + SZ;
  _Float16* Vh  = Kh + SZ;
  _Float16* Oh  = Vh + SZ;
  _Float16* VtG = Oh + SZ;
  _Float16* WtG = VtG + SZ;  // 1024*1024 f16, reused for each weight

  dim3 blk(256);
  dim3 gg(8, 64);    // GEMM: N tiles x M tiles
  dim3 gw(16, 16);   // wtrans tiles
  dim3 gv(16, 128);  // vtrans: key tiles x hb

  wtrans<<<gw, blk, 0, stream>>>(Wq, WtG);
  gemm128<float, _Float16><<<gg, blk, 0, stream>>>(x, WtG, bq, Qh);
  wtrans<<<gw, blk, 0, stream>>>(Wk, WtG);
  gemm128<float, _Float16><<<gg, blk, 0, stream>>>(y, WtG, bk, Kh);
  wtrans<<<gw, blk, 0, stream>>>(Wv, WtG);
  gemm128<float, _Float16><<<gg, blk, 0, stream>>>(y, WtG, bv, Vh);
  vtrans<<<gv, blk, 0, stream>>>(Vh, VtG);
  attn128<<<dim3(128, 8), blk, 0, stream>>>(Qh, Kh, VtG, Oh);
  wtrans<<<gw, blk, 0, stream>>>(Wo, WtG);
  gemm128<_Float16, float><<<gg, blk, 0, stream>>>(Oh, WtG, bo, (float*)d_out);
}
